// HierarchicalAttention_34351148434168
// MI455X (gfx1250) — compile-verified, matrix operand order checked by reading, not run on hardware
//
#include <hip/hip_runtime.h>
#include <hip/hip_bf16.h>
#include <math.h>

// Problem constants (match reference setup_inputs)
constexpr int B = 32, K = 64, S = 2048, D = 512;
constexpr int NCHUNK = 16, CS = S / NCHUNK;   // s-chunking for the g-reduction pass

// Workspace layout (float offsets). Total ~371K floats (~1.45 MB).
constexpr size_t OFF_SF   = 0;                         // B*K : v_kv . W_f[:D]
constexpr size_t OFF_SI   = OFF_SF   + (size_t)B*K;    // B*K : v_kv . W_i[:D]
constexpr size_t OFF_CTXF = OFF_SI   + (size_t)B*K;    // B   : ctx . W_f[D:]
constexpr size_t OFF_CTXI = OFF_CTXF + (size_t)B;      // B   : ctx . W_i[D:2D]
constexpr size_t OFF_ATT  = OFF_CTXI + (size_t)B;      // B*K : kv softmax
constexpr size_t OFF_KW   = OFF_ATT  + (size_t)B*K;    // B*S : exp'd intra-field score
constexpr size_t OFF_SEG  = OFF_KW   + (size_t)B*S;    // B*K : segment norms
constexpr size_t OFF_FAK  = OFF_SEG  + (size_t)B*K;    // B*K : fusion_att bin sums
constexpr size_t OFF_U    = OFF_FAK  + (size_t)B*K;    // B*2D: [w | g]
constexpr size_t OFF_GP   = OFF_U    + (size_t)B*2*D;  // B*NCHUNK*D partials

__device__ __forceinline__ float wave_reduce(float v) {
  #pragma unroll
  for (int o = 16; o > 0; o >>= 1) v += __shfl_down(v, o, 32);
  return v;
}

// ---- 1) per-(b,k) dots with W_f/W_i first halves; last B blocks: context dots ----
__global__ void dots_kernel(const float* __restrict__ vkv, const float* __restrict__ ctx,
                            const float* __restrict__ Wf, const float* __restrict__ Wi,
                            float* __restrict__ ws) {
  __shared__ float red[2][4];
  const int blk = blockIdx.x, tid = threadIdx.x;
  const float *src, *wf, *wi;
  float *of, *oi;
  if (blk < B * K) {
    src = vkv + (size_t)blk * D;
    wf = Wf;     wi = Wi;
    of = ws + OFF_SF + blk;  oi = ws + OFF_SI + blk;
  } else {
    const int b = blk - B * K;
    src = ctx + (size_t)b * D;
    wf = Wf + D; wi = Wi + D;
    of = ws + OFF_CTXF + b;  oi = ws + OFF_CTXI + b;
  }
  float a1 = 0.f, a2 = 0.f;
  for (int d = tid; d < D; d += 128) {
    const float x = src[d];
    a1 += x * wf[d];  a2 += x * wi[d];
  }
  a1 = wave_reduce(a1);  a2 = wave_reduce(a2);
  const int wid = tid >> 5, lane = tid & 31;
  if (lane == 0) { red[0][wid] = a1; red[1][wid] = a2; }
  __syncthreads();
  if (tid == 0) {
    float s1 = 0.f, s2 = 0.f;
    #pragma unroll
    for (int w = 0; w < 4; ++w) { s1 += red[0][w]; s2 += red[1][w]; }
    *of = s1;  *oi = s2;
  }
}

// ---- 2) masked softmax over K per batch ----
__global__ void softmax_kernel(const int* __restrict__ kvnum, float* __restrict__ ws) {
  __shared__ float sh[K];
  const int b = blockIdx.x, k = threadIdx.x;
  const int n = kvnum[b];
  const bool valid = (k < n);
  const float sc = valid ? (ws[OFF_SF + b * K + k] + ws[OFF_CTXF + b]) : -INFINITY;
  sh[k] = sc; __syncthreads();
  for (int o = 32; o > 0; o >>= 1) { if (k < o) sh[k] = fmaxf(sh[k], sh[k + o]); __syncthreads(); }
  const float m = sh[0]; __syncthreads();
  const float e = valid ? expf(sc - m) : 0.f;
  sh[k] = e; __syncthreads();
  for (int o = 32; o > 0; o >>= 1) { if (k < o) sh[k] += sh[k + o]; __syncthreads(); }
  ws[OFF_ATT + b * K + k] = e / sh[0];
}

// ---- 3) stream #1 over f_kwc: kw_score = exp(f_kwc.Wi_f + gathered biases) ----
__global__ void kw_kernel(const float* __restrict__ fkwc, const float* __restrict__ Wi,
                          const int* __restrict__ kvpos, float* __restrict__ ws) {
  __shared__ float w[D];
  const int tid = threadIdx.x;
  for (int d = tid; d < D; d += 256) w[d] = Wi[2 * D + d];
  __syncthreads();
  const int wid = tid >> 5, lane = tid & 31;
  const long row = (long)blockIdx.x * 8 + wid;          // in [0, B*S)
  const float* p = fkwc + (size_t)row * D;
  float acc = 0.f;
  #pragma unroll
  for (int it = 0; it < 4; ++it) {
    const int base = (it * 32 + lane) * 4;
    const float4 v = *(const float4*)(p + base);
    acc += v.x * w[base] + v.y * w[base + 1] + v.z * w[base + 2] + v.w * w[base + 3];
  }
  acc = wave_reduce(acc);
  if (lane == 0) {
    const int b = (int)(row >> 11);
    const int kp = kvpos[row];
    ws[OFF_KW + row] = expf(acc + ws[OFF_SI + b * K + kp] + ws[OFF_CTXI + b]);
  }
}

// ---- 4/6) deterministic segment bin sums: one wave per (b,k) bin ----
__global__ void binsum_kernel(const int* __restrict__ kvpos, const int* __restrict__ mask,
                              const float* __restrict__ src, float* __restrict__ dst,
                              int use_mask) {
  const int tid = threadIdx.x, wid = tid >> 5, lane = tid & 31;
  const int bin = blockIdx.x * 8 + wid;                 // b*K + k
  const int b = bin / K, k = bin % K;
  const int* kp = kvpos + (size_t)b * S;
  const int* mk = mask  + (size_t)b * S;
  const float* sp = src + (size_t)b * S;
  float acc = 0.f;
  for (int s = lane; s < S; s += 32) {
    const bool sel = (kp[s] == k) && (!use_mask || mk[s] != 0);
    if (sel) acc += sp[s];
  }
  acc = wave_reduce(acc);
  if (lane == 0) dst[bin] = acc;
}

// ---- 5) fusion_att = mask ? kv_att * kw_score / seg_norm : 0 ----
__global__ void fusion_kernel(const int* __restrict__ kvpos, const int* __restrict__ mask,
                              const float* __restrict__ ws, float* __restrict__ fa_out) {
  const long i = (long)blockIdx.x * 256 + threadIdx.x;  // b*S + s
  const int b = (int)(i >> 11);
  float v = 0.f;
  if (mask[i]) {
    const int kp = kvpos[i];
    v = ws[OFF_ATT + b * K + kp] * ws[OFF_KW + i] / ws[OFF_SEG + b * K + kp];
  }
  fa_out[i] = v;
}

// ---- 6b) w[b,:] = sum_k fa_k[b,k] * v_kv[b,k,:]  -> U[b, 0:D] ----
__global__ void wsum_kernel(const float* __restrict__ vkv, float* __restrict__ ws) {
  __shared__ float f[K];
  const int b = blockIdx.x, d = threadIdx.x;            // block = 512
  if (d < K) f[d] = ws[OFF_FAK + b * K + d];
  __syncthreads();
  float acc = 0.f;
  const float* vb = vkv + (size_t)b * K * D;
  #pragma unroll 4
  for (int k = 0; k < K; ++k) acc += f[k] * vb[(size_t)k * D + d];
  ws[OFF_U + (size_t)b * 2 * D + d] = acc;
}

// ---- 7) stream #2: chunked partials of g[b,:] = sum_s fa * f_kwc ----
__global__ void gpart_kernel(const float* __restrict__ fkwc, const float* __restrict__ fa,
                             float* __restrict__ ws) {
  const int d = blockIdx.x * 256 + threadIdx.x;
  const int c = blockIdx.y, b = blockIdx.z;
  const float* fb = fa + (size_t)b * S + (size_t)c * CS;
  const float* pb = fkwc + ((size_t)b * S + (size_t)c * CS) * D;
  float acc = 0.f;
  for (int s = 0; s < CS; ++s) acc += fb[s] * pb[(size_t)s * D + d];
  ws[OFF_GP + ((size_t)b * NCHUNK + c) * D + d] = acc;
}

__global__ void greduce_kernel(float* __restrict__ ws) {
  const int b = blockIdx.x, d = threadIdx.x;            // block = 512
  float acc = 0.f;
  #pragma unroll
  for (int c = 0; c < NCHUNK; ++c) acc += ws[OFF_GP + ((size_t)b * NCHUNK + c) * D + d];
  ws[OFF_U + (size_t)b * 2 * D + D + d] = acc;
}

// ---- 8) WMMA GEMM: field_value(32x512) = U(32x1024) @ W_v^T(1024x512) ----
typedef __attribute__((ext_vector_type(2)))  float    v2f;
typedef __attribute__((ext_vector_type(8)))  float    v8f;
typedef __attribute__((ext_vector_type(16))) _Float16 v16h;

__global__ void wmma_out_kernel(const float* __restrict__ Wv, const float* __restrict__ ws,
                                float* __restrict__ out) {
  const int lane = threadIdx.x;           // one full wave32 per 16x16 tile (EXEC all ones)
  const int n0 = blockIdx.x * 16;         // output col tile (h)
  const int m0 = blockIdx.y * 16;         // output row tile (b)
  const int half = lane >> 4, l = lane & 15;
  const float* U = ws + OFF_U;            // B x 2D row-major
  v8f c = {};
#if defined(__has_builtin) && __has_builtin(__builtin_amdgcn_wmma_f32_16x16x4_f32)
  for (int e0 = 0; e0 < 2 * D; e0 += 4) {
    // A tile 16x4: lanes 0-15 hold K={0,1}, lanes 16-31 K={2,3} for row M=l
    const float* ap = U  + (size_t)(m0 + l) * (2 * D) + e0 + 2 * half;
    // B tile 4x16: B[e][n] = Wv[n][e]; mirrored K split across lane halves
    const float* bp = Wv + (size_t)(n0 + l) * (2 * D) + e0 + 2 * half;
    v2f a;  a.x  = ap[0]; a.y  = ap[1];
    v2f bb; bb.x = bp[0]; bb.y = bp[1];
    c = __builtin_amdgcn_wmma_f32_16x16x4_f32(false, a, false, bb, (short)0, c, false, false);
  }
#else
  for (int e0 = 0; e0 < 2 * D; e0 += 32) {
    const float* ap = U  + (size_t)(m0 + l) * (2 * D) + e0;
    const float* bp = Wv + (size_t)(n0 + l) * (2 * D) + e0;
    const int k1 = 8 * half, k2 = 16 + 8 * half;      // 16-bit A/B K-split per ISA layout
    v16h a, bb;
    #pragma unroll
    for (int e = 0; e < 8; ++e) {
      a[e]      = (_Float16)ap[k1 + e];  a[8 + e]  = (_Float16)ap[k2 + e];
      bb[e]     = (_Float16)bp[k1 + e];  bb[8 + e] = (_Float16)bp[k2 + e];
    }
    c = __builtin_amdgcn_wmma_f32_16x16x32_f16(false, a, false, bb, (short)0, c, false, false);
  }
#endif
  // C/D layout: VGPR r -> row M = 8*half + r, col N = l
  #pragma unroll
  for (int r = 0; r < 8; ++r)
    out[(size_t)(m0 + 8 * half + r) * D + (n0 + l)] = c[r];
}

extern "C" void kernel_launch(void* const* d_in, const int* in_sizes, int n_in,
                              void* d_out, int out_size, void* d_ws, size_t ws_size,
                              hipStream_t stream) {
  const float* context = (const float*)d_in[0];
  const float* v_kv    = (const float*)d_in[1];
  const float* f_kwc   = (const float*)d_in[2];
  const float* W_f     = (const float*)d_in[3];
  const float* W_i     = (const float*)d_in[4];
  const float* W_v     = (const float*)d_in[5];
  const int*   kv_pos  = (const int*)d_in[6];
  const int*   kv_num  = (const int*)d_in[7];
  const int*   mask    = (const int*)d_in[8];
  float* out_fv = (float*)d_out;          // (1,B,D) -> B*D floats
  float* out_fa = out_fv + B * D;         // (B,S)
  float* ws = (float*)d_ws;
  (void)in_sizes; (void)n_in; (void)out_size; (void)ws_size;

  dots_kernel   <<<B * K + B, 128, 0, stream>>>(v_kv, context, W_f, W_i, ws);
  softmax_kernel<<<B, K, 0, stream>>>(kv_num, ws);
  kw_kernel     <<<(B * S) / 8, 256, 0, stream>>>(f_kwc, W_i, kv_pos, ws);
  binsum_kernel <<<(B * K) / 8, 256, 0, stream>>>(kv_pos, mask, ws + OFF_KW, ws + OFF_SEG, 1);
  fusion_kernel <<<(B * S) / 256, 256, 0, stream>>>(kv_pos, mask, ws, out_fa);
  binsum_kernel <<<(B * K) / 8, 256, 0, stream>>>(kv_pos, mask, out_fa, ws + OFF_FAK, 0);
  wsum_kernel   <<<B, D, 0, stream>>>(v_kv, ws);
  gpart_kernel  <<<dim3(D / 256, NCHUNK, B), 256, 0, stream>>>(f_kwc, out_fa, ws);
  greduce_kernel<<<B, D, 0, stream>>>(ws);
  wmma_out_kernel<<<dim3(D / 16, B / 16), 32, 0, stream>>>(W_v, ws, out_fv);
}
